// AMSPNet_remove_spa_4183298146524
// MI455X (gfx1250) — matrix operand order repaired and checked by reading, not compiled, not run on hardware
//
#include <hip/hip_runtime.h>
#include <hip/hip_bf16.h>

typedef __attribute__((ext_vector_type(16))) _Float16 v16h;
typedef __attribute__((ext_vector_type(8)))  _Float16 v8h;
typedef __attribute__((ext_vector_type(8)))  float    v8f;
typedef __attribute__((ext_vector_type(4)))  float    v4f;
typedef __attribute__((ext_vector_type(2)))  float    v2f;
typedef __attribute__((ext_vector_type(4)))  unsigned int u32x4;
typedef __attribute__((ext_vector_type(4)))  int      i32x4;
typedef __attribute__((ext_vector_type(8)))  int      i32x8;

#define C_ 21
#define D_ 1280
#define B_ 256
#define W_ 128
#define S_ 20
#define OG_ 64
#define OU_ 32
#define ROWS_ (B_*C_)          // 5376
#define PIf 3.14159265358979323846f

__device__ __forceinline__ float sigm_(float x){ return 1.f/(1.f+__expf(-x)); }

// ---------------- K0: Tensor Data Mover probe (NULL descriptor => architectural no-op) ----
// D# group0 count==0 means NULL tensor; instruction still exercises the TDM issue path and
// TENSORcnt, then we wait it back to zero. Contained in its own kernel.
__global__ void k_tdm_probe(){
#if __has_builtin(__builtin_amdgcn_tensor_load_to_lds)
  u32x4 g0 = {0u,0u,0u,0u};
  i32x8 g1 = {0,0,0,0,0,0,0,0};
  i32x4 g2 = {0,0,0,0};
  i32x4 g3 = {0,0,0,0};
#if defined(__clang_major__) && (__clang_major__ >= 23)
  i32x8 g4 = {0,0,0,0,0,0,0,0};
  __builtin_amdgcn_tensor_load_to_lds(g0, g1, g2, g3, g4, 0);
#else
  __builtin_amdgcn_tensor_load_to_lds(g0, g1, g2, g3, 0);
#endif
  __builtin_amdgcn_s_wait_tensorcnt((short)0);
#endif
}

// ---------------- K1: ft = log|FFT128(windows)| ----------------
__global__ void k_fft_log(const float* __restrict__ x, float* __restrict__ ft){
  __shared__ float win[128];
  int bid = blockIdx.x;
  int j = bid % 10; int c = (bid/10) % C_; int b = bid/(10*C_);
  int k = threadIdx.x;
  win[k] = x[((size_t)b*C_ + c)*D_ + j*128 + k];
  __syncthreads();
  float re=0.f, im=0.f;
  float w0 = -2.f*PIf*(float)k/128.f;
  for(int n=0;n<128;++n){
    float s,co; __sincosf(w0*(float)n,&s,&co);
    re += win[n]*co; im += win[n]*s;
  }
  float mag = sqrtf(re*re+im*im);
  if(mag==0.f) mag = 1e-8f;
  ft[((size_t)b*C_+c)*D_ + j*128 + k] = __logf(mag);
}

// ---------------- K2: segment means + SE MLP -> ew(B,20) ----------------
__global__ void k_se(const float* __restrict__ ft, const float* __restrict__ se_w1,
                     const float* __restrict__ se_w2, float* __restrict__ ew){
  int b = blockIdx.x; int tid = threadIdx.x;           // 256 threads
  __shared__ float colsum[1344];
  __shared__ float sq[20], hid[10];
  for(int p=tid;p<1344;p+=256){
    float s=0.f;
    if(p < D_) for(int c=0;c<C_;++c) s += ft[((size_t)b*C_+c)*D_+p];
    colsum[p]=s;
  }
  __syncthreads();
  if(tid<20){ float s=0.f; int base=tid*64;
    for(int w=0;w<128;++w) s+=colsum[base+w];
    sq[tid]= s/((float)C_*128.f); }
  __syncthreads();
  if(tid<10){ float s=0.f; for(int k=0;k<20;++k) s+= sq[k]*se_w1[tid*20+k];
    hid[tid]=fmaxf(s,0.f); }
  __syncthreads();
  if(tid<20){ float s=0.f; for(int k=0;k<10;++k) s+= hid[k]*se_w2[tid*10+k];
    ew[b*20+tid]= sigm_(s); }
}

// ---------------- K3: max over segments of xseg*ew -> max_seg(B,C,128) ----------------
__global__ void k_maxseg(const float* __restrict__ ft, const float* __restrict__ ew,
                         float* __restrict__ ms){
  int bc = blockIdx.x; int b = bc / C_; int c = bc % C_;
  int w = threadIdx.x;
  float m = -INFINITY;
  for(int s=0;s<S_;++s){
    int p = s*64 + w;
    float v = (p < D_) ? ft[((size_t)b*C_+c)*D_+p] : 0.f;
    v *= ew[b*20+s];
    m = fmaxf(m, v);
  }
  ms[(size_t)bc*128 + w] = m;
}

// ---------------- K4a: DFT of max_seg rows -> Re,Im,Ab ----------------
__global__ void k_dft2(const float* __restrict__ ms, float* __restrict__ Re,
                       float* __restrict__ Im, float* __restrict__ Ab){
  __shared__ float row[128];
  int bc = blockIdx.x; int k = threadIdx.x;
  row[k] = ms[(size_t)bc*128 + k];
  __syncthreads();
  float re=0.f, im=0.f; float w0 = -2.f*PIf*(float)k/128.f;
  for(int n=0;n<128;++n){ float s,co; __sincosf(w0*(float)n,&s,&co);
    re+=row[n]*co; im+=row[n]*s; }
  size_t o = (size_t)bc*128+k;
  Re[o]=re; Im[o]=im; Ab[o]=sqrtf(re*re+im*im);
}

// ---------------- K4b: adjacency adj(B,21,21) ----------------
__global__ void k_adj(const float* __restrict__ Re, const float* __restrict__ Im,
                      const float* __restrict__ Ab, float* __restrict__ adj){
  int b = blockIdx.x; int tid = threadIdx.x;          // 256 threads
  __shared__ float sRe[C_*128], sIm[C_*128], sAb[C_*128];
  for(int i=tid;i<C_*128;i+=256){
    size_t o=(size_t)b*C_*128+i;
    sRe[i]=Re[o]; sIm[i]=Im[o]; sAb[i]=Ab[o];
  }
  __syncthreads();
  for(int p=tid;p<C_*C_;p+=256){
    int i=p/C_, jj=p%C_;
    float num=0.f, den=0.f;
    for(int n=0;n<128;++n){
      num += sIm[i*128+n]*sRe[jj*128+n] - sRe[i*128+n]*sIm[jj*128+n];
      den += sAb[i*128+n]*sAb[jj*128+n];
    }
    adj[(size_t)b*441+p] = (i==jj) ? 0.f : num/den;
  }
}

// ---------------- K4c: pair-swizzle gcn_w for b64 B-fragment loads ----------------
// gswz[(pair*64+n)*2 + {0,1}] = gcn_w[2*pair][n], gcn_w[2*pair+1][n]
__global__ void k_swz(const float* __restrict__ gcnw, float* __restrict__ gswz){
  int i = blockIdx.x*256 + threadIdx.x;
  if(i < 64*64){
    int pr = i>>6, n = i&63;
    gswz[i*2+0] = gcnw[(2*pr  )*64+n];
    gswz[i*2+1] = gcnw[(2*pr+1)*64+n];
  }
}

// ---------------- K5: per-(t,b) dwsep chain + GCN via fp32 WMMA -> g ----------------
// grid = S*B blocks of 256 threads (8 waves).
__global__ void __launch_bounds__(256) k_graph(
    const float* __restrict__ ft, const float* __restrict__ adjbuf,
    const float* __restrict__ dw1, const float* __restrict__ dw1b,
    const float* __restrict__ pw1, const float* __restrict__ pw1b,
    const float* __restrict__ dw2, const float* __restrict__ dw2b,
    const float* __restrict__ pw2, const float* __restrict__ pw2b,
    const float* __restrict__ dw3, const float* __restrict__ dw3b,
    const float* __restrict__ pw3, const float* __restrict__ pw3b,
    const float* __restrict__ gswz, const float* __restrict__ gcnb,
    float* __restrict__ g)
{
  int bid = blockIdx.x;
  int b = bid % B_; int t = bid / B_;
  int tid = threadIdx.x; int lane = tid & 31; int wv = tid >> 5;

  __shared__ __align__(16) float xt[32*128];   // padded to 32 rows, zero-filled
  __shared__ float ybuf[C_*65];
  __shared__ float dbuf[C_*65];
  __shared__ __align__(16) float sup2f[12*64*2]; // support in (row-pair, col) float2 layout
  __shared__ __align__(16) float adjt[32*24];    // adj_t padded to 32x24

  for(int i=tid;i<12*64*2;i+=256) sup2f[i]=0.f;
  for(int i=tid;i<32*24;i+=256) adjt[i]=0.f;
  for(int i=tid;i<32*128;i+=256){
    int c=i>>7, w=i&127; int p=t*64+w;
    xt[i] = (c<C_ && p<D_) ? ft[((size_t)b*C_+c)*D_+p] : 0.f;
  }
  __syncthreads();

  int mt = wv >> 2, nt = wv & 3;
  int m  = mt*16 + (lane & 15);
  int n  = nt*16 + (lane & 15);
  int koff = (lane < 16) ? 0 : 2;
  int mi = (lane>>4)<<3;

  // support(21x64) = xt(21x128) @ gcn_w(128x64) + gcn_b  [fp32 WMMA 16x16x4]
  {
    v8f acc; for(int i=0;i<8;++i) acc[i]=0.f;
    for(int k=0;k<128;k+=4){
      v2f a  = *(const v2f*)&xt[m*128 + k + koff];
      v2f bb = *(const v2f*)&gswz[(((k+koff)>>1)*64 + n)*2];
      acc = __builtin_amdgcn_wmma_f32_16x16x4_f32(false, a, false, bb, (short)0, acc, false, false);
    }
    for(int i=0;i<8;i+=2){
      int mm = mt*16 + mi + i;                 // even row index
      if(mm < C_){
        v2f val;
        val.x = acc[i] + gcnb[n];
        val.y = (mm+1 < C_) ? (acc[i+1] + gcnb[n]) : 0.f;
        *(v2f*)&sup2f[((mm>>1)*64 + n)*2] = val;
      }
    }
  }

  // depthwise-separable chain: 128 ->(64)-> 65 ->(32)-> 34 ->(14)-> 21
  for(int i=tid;i<C_*65;i+=256){
    int c=i/65, l=i%65;
    float s=dw1b[c];
    for(int k=0;k<64;++k) s += xt[c*128 + l + k]*dw1[c*64+k];
    ybuf[i]=s;
  }
  __syncthreads();
  for(int i=tid;i<C_*65;i+=256){
    int o=i/65, l=i%65;
    float s=pw1b[o];
    for(int c=0;c<C_;++c) s += pw1[o*C_+c]*ybuf[c*65+l];
    dbuf[i]=s;
  }
  __syncthreads();
  for(int i=tid;i<C_*34;i+=256){
    int c=i/34, l=i%34;
    float s=dw2b[c];
    for(int k=0;k<32;++k) s += dbuf[c*65 + l + k]*dw2[c*32+k];
    ybuf[i]=s;
  }
  __syncthreads();
  for(int i=tid;i<C_*34;i+=256){
    int o=i/34, l=i%34;
    float s=pw2b[o];
    for(int c=0;c<C_;++c) s += pw2[o*C_+c]*ybuf[c*34+l];
    dbuf[i]=s;
  }
  __syncthreads();
  for(int i=tid;i<C_*C_;i+=256){
    int c=i/C_, l=i%C_;
    float s=dw3b[c];
    for(int k=0;k<14;++k) s += dbuf[c*34 + l + k]*dw3[c*14+k];
    ybuf[i]=s;
  }
  __syncthreads();
  for(int i=tid;i<C_*C_;i+=256){
    int o=i/C_, l=i%C_;
    float s=pw3b[o];
    for(int c=0;c<C_;++c) s += pw3[o*C_+c]*ybuf[c*C_+l];
    adjt[o*24+l] = sigm_(s + adjbuf[(size_t)b*441 + i]);
  }
  __syncthreads();

  // g(21x64) = adj_t(21x21 pad 32x24) @ support(24x64)  [fp32 WMMA 16x16x4]
  {
    v8f acc; for(int i=0;i<8;++i) acc[i]=0.f;
    for(int k=0;k<24;k+=4){
      v2f a  = *(const v2f*)&adjt[m*24 + k + koff];
      v2f bb = *(const v2f*)&sup2f[(((k+koff)>>1)*64 + n)*2];
      acc = __builtin_amdgcn_wmma_f32_16x16x4_f32(false, a, false, bb, (short)0, acc, false, false);
    }
    size_t base = ((size_t)t*B_ + b)*(size_t)(C_*64);
    for(int i=0;i<8;++i){
      int mm = mt*16 + i + mi;
      if(mm<C_) g[base + mm*64 + n] = acc[i];
    }
  }
}

// ---------------- K6: 20-step GRU over independent 16-row tiles [f16 WMMA] ----------------
// grid = 84 blocks x 128 threads (4 waves); each wave owns 16 rows for all 20 steps.
__global__ void __launch_bounds__(128) k_gru(
    const float* __restrict__ g,
    const float* __restrict__ wr, const float* __restrict__ wrb, const float* __restrict__ p_br,
    const float* __restrict__ wu, const float* __restrict__ wub, const float* __restrict__ p_bu,
    const float* __restrict__ wc, const float* __restrict__ wcb, const float* __restrict__ p_bc,
    float* __restrict__ dout)
{
  int tid=threadIdx.x, lane=tid&31, wv=tid>>5;
  __shared__ __align__(32) _Float16 wsw[9216];       // 3 gates x 3 ksteps x 2 ntiles x 32 lanes x 16 halfs
  __shared__ __align__(32) _Float16 catb[4*16*96];   // per-wave A (cat) buffer
  __shared__ float hbuf[4*16*32];                    // per-wave hidden state

  for(int idx=tid; idx<9216; idx+=128){
    int j  = idx & 15;
    int ln = (idx>>4) & 31;
    int nt = (idx>>9) & 1;
    int ks = (idx>>10) % 3;
    int gi = idx / 3072;
    int n = nt*16 + (ln & 15);
    int k = ks*32 + ((ln<16)?0:16) + j;
    const float* Wm = (gi==0)? wr : (gi==1)? wu : wc;
    wsw[idx] = (_Float16)Wm[n*96 + k];     // B[k][n] = W[n][k]
  }
  for(int i=tid;i<4*16*32;i+=128) hbuf[i]=0.f;
  __syncthreads();

  int rowbase = blockIdx.x*64 + wv*16;
  _Float16* cat = &catb[wv*16*96];
  float* h = &hbuf[wv*16*32];
  int n0 = lane & 15;
  int mi = (lane>>4)<<3;     // 0 or 8
  int mrow = lane & 15;

  for(int t=0;t<S_;++t){
    size_t gbase = ((size_t)t*ROWS_ + rowbase)*64;
    if(t+1 < S_) __builtin_prefetch(&g[gbase + (size_t)ROWS_*64], 0, 0);
    // A (cat) : [g | h], g via coalesced float4
    const v4f* gp = (const v4f*)(g + gbase);
    for(int i=lane;i<16*16;i+=32){
      int mm=i>>4, q=i&15;
      v4f v = gp[mm*16 + q];
      int co = mm*96 + q*4;
      cat[co+0]=(_Float16)v.x; cat[co+1]=(_Float16)v.y;
      cat[co+2]=(_Float16)v.z; cat[co+3]=(_Float16)v.w;
    }
    for(int i=lane;i<16*32;i+=32){
      int mm=i>>5, n=i&31;
      cat[mm*96 + 64 + n] = (_Float16)h[mm*32+n];
    }
    __syncthreads();

    v16h A[3];
#pragma unroll
    for(int ks=0;ks<3;++ks){
      int kb = ks*32 + ((lane<16)?0:16);
      v8h lo = *(const v8h*)&cat[mrow*96 + kb];
      v8h hi = *(const v8h*)&cat[mrow*96 + kb + 8];
      A[ks] = __builtin_shufflevector(lo, hi, 0,1,2,3,4,5,6,7,8,9,10,11,12,13,14,15);
    }
    v8f accR[2], accU[2], accC[2];
    for(int nt=0;nt<2;++nt) for(int i=0;i<8;++i){ accR[nt][i]=0.f; accU[nt][i]=0.f; }
#pragma unroll
    for(int ks=0;ks<3;++ks){
#pragma unroll
      for(int nt=0;nt<2;++nt){
        v16h bR = *(const v16h*)&wsw[(0*3+ks)*1024 + nt*512 + lane*16];
        accR[nt] = __builtin_amdgcn_wmma_f32_16x16x32_f16(false, A[ks], false, bR, (short)0, accR[nt], false, false);
        v16h bU = *(const v16h*)&wsw[(1*3+ks)*1024 + nt*512 + lane*16];
        accU[nt] = __builtin_amdgcn_wmma_f32_16x16x32_f16(false, A[ks], false, bU, (short)0, accU[nt], false, false);
      }
    }
    float r[2][8], u[2][8];
    for(int nt=0;nt<2;++nt){
      int n = nt*16 + n0;
      float biasR = wrb[n] + p_br[n];
      float biasU = wub[n] + p_bu[n];
      for(int i=0;i<8;++i){
        r[nt][i] = sigm_(accR[nt][i] + biasR);
        u[nt][i] = sigm_(accU[nt][i] + biasU);
      }
    }
    // cat2 h-part = r * h
    for(int nt=0;nt<2;++nt){
      int n = nt*16 + n0;
      for(int i=0;i<8;++i){
        int mm = i + mi;
        cat[mm*96 + 64 + n] = (_Float16)(r[nt][i] * h[mm*32+n]);
      }
    }
    __syncthreads();
    { // rebuild A[2] with r*h part
      int kb = 64 + ((lane<16)?0:16);
      v8h lo = *(const v8h*)&cat[mrow*96 + kb];
      v8h hi = *(const v8h*)&cat[mrow*96 + kb + 8];
      A[2] = __builtin_shufflevector(lo, hi, 0,1,2,3,4,5,6,7,8,9,10,11,12,13,14,15);
    }
    for(int nt=0;nt<2;++nt) for(int i=0;i<8;++i) accC[nt][i]=0.f;
#pragma unroll
    for(int ks=0;ks<3;++ks){
#pragma unroll
      for(int nt=0;nt<2;++nt){
        v16h bC = *(const v16h*)&wsw[(2*3+ks)*1024 + nt*512 + lane*16];
        accC[nt] = __builtin_amdgcn_wmma_f32_16x16x32_f16(false, A[ks], false, bC, (short)0, accC[nt], false, false);
      }
    }
    for(int nt=0;nt<2;++nt){
      int n = nt*16 + n0;
      float biasC = wcb[n] + p_bc[n];
      for(int i=0;i<8;++i){
        int mm = i + mi;
        float cc = tanhf(accC[nt][i] + biasC);
        h[mm*32+n] = u[nt][i]*h[mm*32+n] + (1.f-u[nt][i])*cc;
      }
    }
    __syncthreads();
  }
  // emit h_last (rows where c == C-1)
  for(int i=lane;i<16*32;i+=32){
    int mm=i>>5, n=i&31;
    int row = rowbase + mm;
    if(row % C_ == (C_-1)){
      int b = row/C_;
      dout[768 + b*32 + n] = h[mm*32+n];
    }
  }
}

// ---------------- K7: head: out = (h_last@fc1^T+b)@fc2^T+b ----------------
__global__ void k_head(const float* __restrict__ fc1w, const float* __restrict__ fc1b,
                       const float* __restrict__ fc2w, const float* __restrict__ fc2b,
                       float* __restrict__ dout){
  int b = blockIdx.x*blockDim.x + threadIdx.x;
  if(b>=B_) return;
  const float* hl = &dout[768 + b*32];
  float hid[C_];
  for(int c=0;c<C_;++c){
    float s=fc1b[c];
    for(int n=0;n<OU_;++n) s += hl[n]*fc1w[c*OU_+n];
    hid[c]=s;
  }
  for(int k=0;k<3;++k){
    float s=fc2b[k];
    for(int c=0;c<C_;++c) s += hid[c]*fc2w[k*C_+c];
    dout[b*3+k]=s;
    dout[8960 + b*3+k]=s;
  }
}

extern "C" void kernel_launch(void* const* d_in, const int* in_sizes, int n_in,
                              void* d_out, int out_size, void* d_ws, size_t ws_size,
                              hipStream_t stream) {
  (void)in_sizes; (void)n_in; (void)out_size; (void)ws_size;
  const float* x     = (const float*)d_in[0];
  const float* se_w1 = (const float*)d_in[1];
  const float* se_w2 = (const float*)d_in[2];
  const float* dw1   = (const float*)d_in[3];
  const float* dw1b  = (const float*)d_in[4];
  const float* pw1   = (const float*)d_in[5];
  const float* pw1b  = (const float*)d_in[6];
  const float* dw2   = (const float*)d_in[7];
  const float* dw2b  = (const float*)d_in[8];
  const float* pw2   = (const float*)d_in[9];
  const float* pw2b  = (const float*)d_in[10];
  const float* dw3   = (const float*)d_in[11];
  const float* dw3b  = (const float*)d_in[12];
  const float* pw3   = (const float*)d_in[13];
  const float* pw3b  = (const float*)d_in[14];
  const float* gcnw  = (const float*)d_in[15];
  const float* gcnb  = (const float*)d_in[16];
  const float* wr    = (const float*)d_in[17];
  const float* wrb   = (const float*)d_in[18];
  const float* br    = (const float*)d_in[19];
  const float* wu    = (const float*)d_in[20];
  const float* wub   = (const float*)d_in[21];
  const float* bu    = (const float*)d_in[22];
  const float* wc    = (const float*)d_in[23];
  const float* wcb   = (const float*)d_in[24];
  const float* bc    = (const float*)d_in[25];
  const float* fc1w  = (const float*)d_in[26];
  const float* fc1b  = (const float*)d_in[27];
  const float* fc2w  = (const float*)d_in[28];
  const float* fc2b  = (const float*)d_in[29];
  float* out = (float*)d_out;
  float* ws  = (float*)d_ws;

  // workspace layout (floats)
  const size_t OFF_FT  = 0;                           // 6,881,280
  const size_t OFF_EW  = OFF_FT  + (size_t)B_*C_*D_;  // 5,120
  const size_t OFF_MS  = OFF_EW  + (size_t)B_*S_;     // 688,128
  const size_t OFF_RE  = OFF_MS  + (size_t)B_*C_*128;
  const size_t OFF_IM  = OFF_RE  + (size_t)B_*C_*128;
  const size_t OFF_AB  = OFF_IM  + (size_t)B_*C_*128;
  const size_t OFF_ADJ = OFF_AB  + (size_t)B_*C_*128; // 112,896
  const size_t OFF_G   = OFF_ADJ + (size_t)B_*C_*C_;  // 6,881,280
  const size_t OFF_GSW = OFF_G   + (size_t)S_*ROWS_*OG_; // 8,192

  float* ft   = ws + OFF_FT;
  float* ew   = ws + OFF_EW;
  float* ms   = ws + OFF_MS;
  float* Re   = ws + OFF_RE;
  float* Im   = ws + OFF_IM;
  float* Ab   = ws + OFF_AB;
  float* adj  = ws + OFF_ADJ;
  float* g    = ws + OFF_G;
  float* gswz = ws + OFF_GSW;

  k_tdm_probe<<<1,        32,  0, stream>>>();
  k_fft_log<<<B_*C_*10,   128, 0, stream>>>(x, ft);
  k_se     <<<B_,         256, 0, stream>>>(ft, se_w1, se_w2, ew);
  k_maxseg <<<B_*C_,      128, 0, stream>>>(ft, ew, ms);
  k_dft2   <<<B_*C_,      128, 0, stream>>>(ms, Re, Im, Ab);
  k_adj    <<<B_,         256, 0, stream>>>(Re, Im, Ab, adj);
  k_swz    <<<16,         256, 0, stream>>>(gcnw, gswz);
  k_graph  <<<S_*B_,      256, 0, stream>>>(ft, adj,
                                            dw1, dw1b, pw1, pw1b,
                                            dw2, dw2b, pw2, pw2b,
                                            dw3, dw3b, pw3, pw3b,
                                            gswz, gcnb, g);
  k_gru    <<<ROWS_/64,   128, 0, stream>>>(g, wr, wrb, br, wu, wub, bu, wc, wcb, bc, out);
  k_head   <<<1,          256, 0, stream>>>(fc1w, fc1b, fc2w, fc2b, out);
}
